// Attention_5282809774979
// MI455X (gfx1250) — matrix operand order repaired and checked by reading, not compiled, not run on hardware
//
#include <hip/hip_runtime.h>
#include <hip/hip_bf16.h>

#define BG 16
#define Nq 2048
#define Dd 512

typedef __attribute__((ext_vector_type(16))) __bf16 bf16x16;
typedef __attribute__((ext_vector_type(8)))  float  f32x8;
typedef unsigned int u32x4 __attribute__((ext_vector_type(4)));
typedef int          i32x4 __attribute__((ext_vector_type(4)));
typedef int          i32x8 __attribute__((ext_vector_type(8)));

union BfOp { bf16x16 v; uint4 q[2]; unsigned short s[16]; __bf16 h[16]; };

__device__ __forceinline__ unsigned short f2bf(float f) {
  __bf16 h = (__bf16)f;            // lowers to v_cvt_pk_bf16_f32
  return __builtin_bit_cast(unsigned short, h);
}

__device__ __forceinline__ f32x8 wmma_bf16(bf16x16 a, bf16x16 b, f32x8 c) {
  // (neg_a, A, neg_b, B, c_mod, C, reuse_a, reuse_b)
  return __builtin_amdgcn_wmma_f32_16x16x32_bf16(false, a, false, b, (short)0, c,
                                                 false, false);
}

// ---- convert pr to bf16 + build transposed copy (tiled through LDS) --------
__global__ __launch_bounds__(256) void k_conv_pr(const float* __restrict__ pr,
                                                 unsigned short* __restrict__ prb,
                                                 unsigned short* __restrict__ prTb) {
  __shared__ unsigned short tile[64][72];
  int bid = blockIdx.x;
  int bg = bid >> 8;                 // 256 tiles (32 x 8) per bg
  int rr = bid & 255;
  int n0 = (rr >> 3) << 6;
  int d0 = (rr & 7) << 6;
  int tid = threadIdx.x;
#pragma unroll
  for (int i = 0; i < 16; ++i) {
    int e = tid + i * 256;
    int row = e >> 6, col = e & 63;
    size_t gi = (size_t)(bg * Nq + n0 + row) * Dd + d0 + col;
    unsigned short us = f2bf(pr[gi]);
    prb[gi] = us;
    tile[row][col] = us;
  }
  __syncthreads();
#pragma unroll
  for (int i = 0; i < 16; ++i) {
    int e = tid + i * 256;
    int dr = e >> 6, nc = e & 63;
    prTb[(size_t)(bg * Dd + d0 + dr) * Nq + n0 + nc] = tile[nc][dr];
  }
}

// ---- M -> bf16 transposed ---------------------------------------------------
__global__ __launch_bounds__(256) void k_conv_M(const float* __restrict__ M,
                                                unsigned short* __restrict__ MTb) {
  __shared__ unsigned short tile[64][72];
  int d0 = (blockIdx.x >> 3) << 6;
  int e0 = (blockIdx.x & 7) << 6;
  int tid = threadIdx.x;
#pragma unroll
  for (int i = 0; i < 16; ++i) {
    int e = tid + i * 256;
    int row = e >> 6, col = e & 63;
    tile[row][col] = f2bf(M[(d0 + row) * Dd + e0 + col]);
  }
  __syncthreads();
#pragma unroll
  for (int i = 0; i < 16; ++i) {
    int e = tid + i * 256;
    int er = e >> 6, dc = e & 63;
    MTb[(e0 + er) * Dd + d0 + dc] = tile[dc][er];
  }
}

// ---- Q = po @ M  (bf16 WMMA, f32 accum, bf16 result) -----------------------
__global__ __launch_bounds__(256) void k_gemmQ(const float* __restrict__ po,
                                               const unsigned short* __restrict__ MTb,
                                               unsigned short* __restrict__ Qb) {
  int bg = blockIdx.x >> 7;
  int q0 = (blockIdx.x & 127) << 4;
  int tid = threadIdx.x;
  int wave = tid >> 5, lane = tid & 31;
  int l16 = lane & 15, lhi = lane >> 4;
  const float* poRow = po + (size_t)(bg * Nq + q0 + l16) * Dd;
  f32x8 acc[4] = {};
#pragma unroll 1
  for (int kt = 0; kt < 16; ++kt) {
    int d0 = kt * 32;
    int kbA = d0 + lhi * 16;         // A layout: halves = K kbA..kbA+15, row = l16
    float4 f0 = *(const float4*)(poRow + kbA);
    float4 f1 = *(const float4*)(poRow + kbA + 4);
    float4 f2 = *(const float4*)(poRow + kbA + 8);
    float4 f3 = *(const float4*)(poRow + kbA + 12);
    BfOp a;
    float tf[16] = {f0.x,f0.y,f0.z,f0.w, f1.x,f1.y,f1.z,f1.w,
                    f2.x,f2.y,f2.z,f2.w, f3.x,f3.y,f3.z,f3.w};
#pragma unroll
    for (int i = 0; i < 16; ++i) a.h[i] = (__bf16)tf[i];
    int K0 = d0 + lhi * 16;          // B layout: halves = K K0..K0+15, col = l16
    BfOp b[4];
#pragma unroll
    for (int t = 0; t < 4; ++t) {
      const uint4* bp = (const uint4*)(MTb + (size_t)(wave * 64 + t * 16 + l16) * Dd + K0);
      b[t].q[0] = bp[0]; b[t].q[1] = bp[1];
    }
#pragma unroll
    for (int t = 0; t < 4; ++t)
      acc[t] = wmma_bf16(a.v, b[t].v, acc[t]);
  }
#pragma unroll
  for (int t = 0; t < 4; ++t) {
    int e0 = wave * 64 + t * 16;
#pragma unroll
    for (int v = 0; v < 8; ++v)      // C/D: M = v + 8*lhi, N = l16
      Qb[(size_t)(bg * Nq + q0 + v + lhi * 8) * Dd + e0 + l16] = f2bf(acc[t][v]);
  }
}

// ---- fused S = Q pr^T, softmax, A write, out = A pr ------------------------
#define QS_STRIDE 520
#define S_STRIDE  2052
#define PB_STRIDE 2056
static constexpr size_t SMEM_ATTN =
    (size_t)16 * QS_STRIDE * 2 + (size_t)16 * S_STRIDE * 4 +
    (size_t)16 * PB_STRIDE * 2 + 256 * 4 + 16 * 4 + 16 * 4;   // 214,912 B

__global__ __launch_bounds__(256) void k_attn(const unsigned short* __restrict__ Qb,
                                              const unsigned short* __restrict__ prb,
                                              const unsigned short* __restrict__ prTb,
                                              float* __restrict__ out,
                                              float* __restrict__ Aout) {
  extern __shared__ __align__(16) unsigned char smem[];
  unsigned short* Qs = (unsigned short*)smem;                                  // 16x520
  float* S  = (float*)(smem + (size_t)16 * QS_STRIDE * 2);                     // 16x2052
  unsigned short* Pb =
      (unsigned short*)(smem + (size_t)16 * QS_STRIDE * 2 + (size_t)16 * S_STRIDE * 4);
  float* red  = (float*)(smem + (size_t)16 * QS_STRIDE * 2 +
                         (size_t)16 * S_STRIDE * 4 + (size_t)16 * PB_STRIDE * 2);
  float* rmax = red + 256;
  float* rsum = rmax + 16;

  int bg = blockIdx.x >> 7;
  int q0 = (blockIdx.x & 127) << 4;
  int tid = threadIdx.x;
  int wave = tid >> 5, lane = tid & 31;
  int l16 = lane & 15, lhi = lane >> 4;

  // --- Stage Q block (16x512 bf16) into LDS via the Tensor Data Mover. ---
  // D#: 16x512 2-byte tile, tensor_dim0_stride=512, LDS padding of 4 dwords
  // after every 256 dwords (one row) -> padded row stride QS_STRIDE=520.
  if (wave == 0) {
    unsigned long long ga =
        (unsigned long long)(Qb + (size_t)(bg * Nq + q0) * Dd);   // byte addr
    u32x4 g0 = { 1u,                                   // count=1, is_restore=0
                 0u,                                   // lds_addr = 0 (dyn LDS base)
                 (unsigned)(ga & 0xffffffffu),
                 (unsigned)((ga >> 32) & 0x01ffffffu) | 0x80000000u }; // type=2
    i32x8 g1 = { (int)((1u << 16) |                    // data_size = 2 bytes
                       (1u << 20) |                    // pad_enable
                       (7u << 22) |                    // pad_interval = 256 dwords
                       (3u << 25)),                    // pad_amount  = 4 dwords
                 (int)(512u << 16),                    // tensor_dim0 = 512
                 (int)(16u << 16),                     // tensor_dim1 = 16
                 (int)(512u << 16),                    // tile_dim0 = 512
                 (int)16,                              // tile_dim1 = 16
                 (int)512,                             // tensor_dim0_stride = 512
                 0, 0 };
    i32x4 g2 = {};
    i32x4 g3 = {};
#if defined(__clang_major__) && (__clang_major__ >= 23)
    i32x8 g4 = {};
    __builtin_amdgcn_tensor_load_to_lds(g0, g1, g2, g3, g4, 0);
#else
    __builtin_amdgcn_tensor_load_to_lds(g0, g1, g2, g3, 0);
#endif
    __builtin_amdgcn_s_wait_tensorcnt(0);
  }
  __syncthreads();

  // Phase 1: scores; double-buffered operands so WMMA group k overlaps the
  // 8 outstanding global_load_b128 of group k+1.
  auto loadA1 = [&](int kt, BfOp& a) {
    int kbA = kt * 32 + lhi * 16;
    a.q[0] = *(const uint4*)(Qs + l16 * QS_STRIDE + kbA);
    a.q[1] = *(const uint4*)(Qs + l16 * QS_STRIDE + kbA + 8);
  };
  auto loadB1 = [&](int kt, int ng, BfOp* b) {
    int K0 = kt * 32 + lhi * 16;
#pragma unroll
    for (int t = 0; t < 4; ++t) {
      int n0 = (wave * 16 + ng * 4 + t) * 16;
      const uint4* bp = (const uint4*)(prb + (size_t)(bg * Nq + n0 + l16) * Dd + K0);
      b[t].q[0] = bp[0]; b[t].q[1] = bp[1];
    }
  };
  for (int ng = 0; ng < 4; ++ng) {
    f32x8 acc[4] = {};
    BfOp a0, a1, b0[4], b1[4];
    loadA1(0, a0); loadB1(0, ng, b0);
#pragma unroll 1
    for (int kt = 0; kt < 14; kt += 2) {
      loadA1(kt + 1, a1); loadB1(kt + 1, ng, b1);
#pragma unroll
      for (int t = 0; t < 4; ++t) acc[t] = wmma_bf16(a0.v, b0[t].v, acc[t]);
      loadA1(kt + 2, a0); loadB1(kt + 2, ng, b0);
#pragma unroll
      for (int t = 0; t < 4; ++t) acc[t] = wmma_bf16(a1.v, b1[t].v, acc[t]);
    }
    loadA1(15, a1); loadB1(15, ng, b1);
#pragma unroll
    for (int t = 0; t < 4; ++t) acc[t] = wmma_bf16(a0.v, b0[t].v, acc[t]);
#pragma unroll
    for (int t = 0; t < 4; ++t) acc[t] = wmma_bf16(a1.v, b1[t].v, acc[t]);
#pragma unroll
    for (int t = 0; t < 4; ++t) {
      int n0 = (wave * 16 + ng * 4 + t) * 16;
#pragma unroll
      for (int v = 0; v < 8; ++v)
        S[(v + lhi * 8) * S_STRIDE + n0 + l16] = acc[t][v];
    }
  }
  __syncthreads();

  // Softmax (block-wide; 16 lanes per row, 128 elems each)
  int r = tid >> 4, j = tid & 15;
  float m = -3.0e38f;
  for (int c = 0; c < 128; ++c) m = fmaxf(m, S[r * S_STRIDE + j * 128 + c]);
  red[r * 16 + j] = m;
  __syncthreads();
  if (j == 0) {
    float mm = red[r * 16];
    for (int k = 1; k < 16; ++k) mm = fmaxf(mm, red[r * 16 + k]);
    rmax[r] = mm;
  }
  __syncthreads();
  float mr = rmax[r], sum = 0.f;
  for (int c = 0; c < 128; ++c) {
    int idx = r * S_STRIDE + j * 128 + c;
    float p = __expf(S[idx] - mr);
    S[idx] = p;
    sum += p;
  }
  red[r * 16 + j] = sum;
  __syncthreads();
  if (j == 0) {
    float ss = red[r * 16];
    for (int k = 1; k < 16; ++k) ss += red[r * 16 + k];
    rsum[r] = 1.0f / ss;
  }
  __syncthreads();

  // normalize; write A (coalesced f32) and bf16 probs to LDS
#pragma unroll 1
  for (int i = 0; i < 128; ++i) {
    int e = tid + i * 256;
    int r2 = e >> 11, c2 = e & 2047;
    float p = S[r2 * S_STRIDE + c2] * rsum[r2];
    Aout[(size_t)(bg * Nq + q0 + r2) * Nq + c2] = p;
    Pb[r2 * PB_STRIDE + c2] = f2bf(p);
  }
  __syncthreads();

  // Phase 3: out = P @ pr (B from pr^T => contiguous K); double-buffered.
  auto loadA3 = [&](int kt, BfOp& a) {
    int kbA = kt * 32 + lhi * 16;
    a.q[0] = *(const uint4*)(Pb + l16 * PB_STRIDE + kbA);
    a.q[1] = *(const uint4*)(Pb + l16 * PB_STRIDE + kbA + 8);
  };
  auto loadB3 = [&](int kt, BfOp* b) {
    int K0 = kt * 32 + lhi * 16;
#pragma unroll
    for (int t = 0; t < 4; ++t) {
      int c0 = wave * 64 + t * 16;
      const uint4* bp = (const uint4*)(prTb + (size_t)(bg * Dd + c0 + l16) * Nq + K0);
      b[t].q[0] = bp[0]; b[t].q[1] = bp[1];
    }
  };
  {
    f32x8 acc[4] = {};
    BfOp a0, a1, b0[4], b1[4];
    loadA3(0, a0); loadB3(0, b0);
#pragma unroll 1
    for (int kt = 0; kt < 62; kt += 2) {
      loadA3(kt + 1, a1); loadB3(kt + 1, b1);
#pragma unroll
      for (int t = 0; t < 4; ++t) acc[t] = wmma_bf16(a0.v, b0[t].v, acc[t]);
      loadA3(kt + 2, a0); loadB3(kt + 2, b0);
#pragma unroll
      for (int t = 0; t < 4; ++t) acc[t] = wmma_bf16(a1.v, b1[t].v, acc[t]);
    }
    loadA3(63, a1); loadB3(63, b1);
#pragma unroll
    for (int t = 0; t < 4; ++t) acc[t] = wmma_bf16(a0.v, b0[t].v, acc[t]);
#pragma unroll
    for (int t = 0; t < 4; ++t) acc[t] = wmma_bf16(a1.v, b1[t].v, acc[t]);
#pragma unroll
    for (int t = 0; t < 4; ++t) {
      int c0 = wave * 64 + t * 16;
#pragma unroll
      for (int v = 0; v < 8; ++v)
        out[(size_t)(bg * Nq + q0 + v + lhi * 8) * Dd + c0 + l16] = acc[t][v];
    }
  }
}

extern "C" void kernel_launch(void* const* d_in, const int* in_sizes, int n_in,
                              void* d_out, int out_size, void* d_ws, size_t ws_size,
                              hipStream_t stream) {
  (void)in_sizes; (void)n_in; (void)out_size; (void)ws_size;
  const float* po = (const float*)d_in[0];
  const float* pr = (const float*)d_in[1];
  const float* M  = (const float*)d_in[2];
  float* out  = (float*)d_out;
  float* Aout = out + (size_t)BG * Nq * Dd;

  const size_t E = (size_t)BG * Nq * Dd;       // 16.7M elems
  unsigned short* Qb  = (unsigned short*)d_ws; // 32 MiB
  unsigned short* prb = Qb + E;                // 32 MiB
  unsigned short* prT = prb + E;               // 32 MiB
  unsigned short* MTb = prT + E;               // 0.5 MiB

  hipFuncSetAttribute(reinterpret_cast<const void*>(k_attn),
                      hipFuncAttributeMaxDynamicSharedMemorySize, (int)SMEM_ATTN);

  k_conv_pr<<<BG * 256, 256, 0, stream>>>(pr, prb, prT);
  k_conv_M<<<64, 256, 0, stream>>>(M, MTb);
  k_gemmQ<<<BG * 128, 256, 0, stream>>>(po, MTb, Qb);
  k_attn<<<BG * 128, 256, SMEM_ATTN, stream>>>(Qb, prb, prT, out, Aout);
}
